// RGCN_DualAttn_FFNN_25297357373584
// MI455X (gfx1250) — compile-verified, matrix operand order checked by reading, not run on hardware
//
#include <hip/hip_runtime.h>
#include <hip/hip_bf16.h>

// ---------------------------------------------------------------------------
// Sizes (match reference)
// ---------------------------------------------------------------------------
#define Nn  50000
#define Rr  8
#define NEe 800000
#define Dd  128
#define Hh  4
#define Bb  4096
#define Ss  50
#define E3  384      // 3*D

typedef __attribute__((ext_vector_type(16))) _Float16 v16h;
typedef __attribute__((ext_vector_type(8)))  float    v8f;

// ---------------------------------------------------------------------------
// Kernel 1: edge scatter.  agg[(dst,rel)] += x[src], cnt[(dst,rel)] += 1.
// One 32-lane group per edge; each lane handles 4 floats (float4 load, 4 fadds).
// ---------------------------------------------------------------------------
__global__ void __launch_bounds__(256)
edge_scatter_kernel(const float* __restrict__ x,
                    const int* __restrict__ src,
                    const int* __restrict__ dst,
                    const int* __restrict__ et,
                    float* __restrict__ agg,
                    float* __restrict__ cnt) {
    int g = blockIdx.x * 256 + threadIdx.x;
    int e = g >> 5;
    int l = g & 31;
    if (e >= NEe) return;
    int s = src[e], d = dst[e], r = et[e];
    long seg = (long)d * Rr + r;
    if (l == 0) unsafeAtomicAdd(cnt + seg, 1.0f);
    const float4 v = *(const float4*)(x + (long)s * Dd + l * 4);
    float* p = agg + seg * Dd + (long)l * 4;
    unsafeAtomicAdd(p + 0, v.x);
    unsafeAtomicAdd(p + 1, v.y);
    unsafeAtomicAdd(p + 2, v.z);
    unsafeAtomicAdd(p + 3, v.w);
}

// ---------------------------------------------------------------------------
// Kernel 2: fused RGCN layer GEMM.
// out[n,:] = sum_r (agg[n,r,:]/cnt[n,r]) @ W_r  +  x_in[n,:] @ W_root + b
// [64-node tile x 1152] @ [1152 x 128] per block.  K loop is split by
// relation segment so the inner loop is branchless; OOB rows are handled by
// index-clamping + zero scale (no exec-mask divergence in the hot loop).
// 256 threads = 8 waves; wave (ms, ng) owns 4 16x16 tiles.
// ---------------------------------------------------------------------------
__global__ void __launch_bounds__(256)
rgcn_gemm_kernel(const float* __restrict__ x_in,
                 const float* __restrict__ agg,
                 const float* __restrict__ cnt,
                 const float* __restrict__ w_rel,    // [8,128,128] (this layer)
                 const float* __restrict__ w_root,   // [128,128]
                 const float* __restrict__ bias,     // [128]
                 float* __restrict__ x_out,
                 int do_relu) {
    __shared__ float inv_s[64][8];
    __shared__ __align__(32) _Float16 Af[4][32][16];   // per-lane fragment layout
    __shared__ __align__(32) _Float16 Bf[8][32][16];

    const int tid  = threadIdx.x;
    const int lane = tid & 31;
    const int w    = tid >> 5;
    const int nbase = blockIdx.x * 64;

    // per-(node,rel) 1/count, staged once (0 for out-of-range rows)
    for (int i = tid; i < 512; i += 256) {
        int row = i >> 3, r = i & 7;
        int n = nbase + row;
        float c = (n < Nn) ? cnt[(long)n * 8 + r] : 0.0f;
        inv_s[row][r] = (c > 0.0f) ? (1.0f / c) : 0.0f;
    }

    const int ms = w & 3;     // M strip (16 rows)
    const int ng = w >> 2;    // N half (64 cols)
    v8f acc0 = {}, acc1 = {}, acc2 = {}, acc3 = {};

    // A staging coords: pair of threads per (mstrip,lane); 8 contiguous k each
    const int pa     = tid >> 1;
    const int a_ms   = pa >> 5;
    const int a_lane = pa & 31;
    const int a_row  = a_ms * 16 + (a_lane & 15);
    const int a_kb   = (a_lane < 16) ? 0 : 8;
    const int a_hb   = (tid & 1) * 8;
    const int a_klo  = a_kb + (a_hb ? 16 : 0);   // ISA A-layout: halves 0-7 -> kb..kb+7, 8-15 -> kb+16..kb+23
    // B staging coords: one thread per (ntile,lane); 16 halves (strided weight reads)
    const int b_nt  = tid >> 5;
    const int b_col = b_nt * 16 + (lane & 15);
    const int b_kb  = (lane < 16) ? 0 : 8;

    const int   n_a    = nbase + a_row;
    const int   nclamp = (n_a < Nn) ? n_a : (Nn - 1);     // always-valid address
    const float valid  = (n_a < Nn) ? 1.0f : 0.0f;        // zero contribution if OOB

    __syncthreads();

    // one 32-wide K step: stage A (scaled f16), stage B, 4 WMMAs
    auto k_step = [&](const float* __restrict__ asrc, float ascale,
                      const float* __restrict__ wcol) {
        #pragma unroll
        for (int j = 0; j < 8; ++j)
            Af[a_ms][a_lane][a_hb + j] = (_Float16)(asrc[j] * ascale);
        #pragma unroll
        for (int j = 0; j < 8; ++j) {
            Bf[b_nt][lane][j]     = (_Float16)wcol[(long)(b_kb + j) * Dd];
            Bf[b_nt][lane][8 + j] = (_Float16)wcol[(long)(b_kb + 16 + j) * Dd];
        }
        __syncthreads();
        v16h a  = *(const v16h*)&Af[ms][lane][0];
        v16h b0 = *(const v16h*)&Bf[ng * 4 + 0][lane][0];
        v16h b1 = *(const v16h*)&Bf[ng * 4 + 1][lane][0];
        v16h b2 = *(const v16h*)&Bf[ng * 4 + 2][lane][0];
        v16h b3 = *(const v16h*)&Bf[ng * 4 + 3][lane][0];
        acc0 = __builtin_amdgcn_wmma_f32_16x16x32_f16(false, a, false, b0, (short)0, acc0, false, false);
        acc1 = __builtin_amdgcn_wmma_f32_16x16x32_f16(false, a, false, b1, (short)0, acc1, false, false);
        acc2 = __builtin_amdgcn_wmma_f32_16x16x32_f16(false, a, false, b2, (short)0, acc2, false, false);
        acc3 = __builtin_amdgcn_wmma_f32_16x16x32_f16(false, a, false, b3, (short)0, acc3, false, false);
        __syncthreads();
    };

    // relation segments 0..7 (A from agg, scaled by 1/cnt)
    for (int rr = 0; rr < Rr; ++rr) {
        // warm GL2 with the next weight panel (gfx1250 global_prefetch path)
        const float* nextw = (rr + 1 < Rr) ? (w_rel + (long)(rr + 1) * Dd * Dd) : w_root;
        __builtin_prefetch(nextw + b_col, 0, 1);
        const float  ascale = inv_s[a_row][rr];           // one LDS read per segment
        const float* abase  = agg + ((long)nclamp * Rr + rr) * Dd + a_klo;
        const float* wbase  = w_rel + (long)rr * Dd * Dd + b_col;
        #pragma unroll
        for (int db = 0; db < Dd; db += 32)
            k_step(abase + db, ascale, wbase + (long)db * Dd);
    }
    // root segment (A from x_in)
    {
        const float* abase = x_in + (long)nclamp * Dd + a_klo;
        const float* wbase = w_root + b_col;
        #pragma unroll
        for (int db = 0; db < Dd; db += 32)
            k_step(abase + db, valid, wbase + (long)db * Dd);
    }

    // C/D layout: lane<16 -> M=vgpr, lane>=16 -> M=vgpr+8; N = lane%16
    const int col0 = ng * 64 + (lane & 15);
    const int mrow = ms * 16 + ((lane < 16) ? 0 : 8);
    #pragma unroll
    for (int i = 0; i < 8; ++i) {
        int n = nbase + mrow + i;
        if (n < Nn) {
            float v0 = acc0[i] + bias[col0 +  0];
            float v1 = acc1[i] + bias[col0 + 16];
            float v2 = acc2[i] + bias[col0 + 32];
            float v3 = acc3[i] + bias[col0 + 48];
            if (do_relu) { v0 = fmaxf(v0,0.f); v1 = fmaxf(v1,0.f); v2 = fmaxf(v2,0.f); v3 = fmaxf(v3,0.f); }
            x_out[(long)n * Dd + col0 +  0] = v0;
            x_out[(long)n * Dd + col0 + 16] = v1;
            x_out[(long)n * Dd + col0 + 32] = v2;
            x_out[(long)n * Dd + col0 + 48] = v3;
        }
    }
}

// ---------------------------------------------------------------------------
// Kernel 3: fused dual attention (query length 1 => no K/V materialization).
// s[h,j] = x_j . (Wk_h^T q_h);  o = Wv . (sum_j a_j x_j) + bv;  res = Wo o + bo
// One 128-thread block per bill.  Writes cat[b] = [left | bill | right].
// ---------------------------------------------------------------------------
__global__ void __launch_bounds__(128)
attn_kernel(const float* __restrict__ x2,
            const int* __restrict__ query_idx,
            const int* __restrict__ spon_idx,
            const int* __restrict__ subj_idx,
            const unsigned char* __restrict__ spon_mask,
            const unsigned char* __restrict__ subj_mask,
            const float* __restrict__ la_in_w,  const float* __restrict__ la_in_b,
            const float* __restrict__ la_out_w, const float* __restrict__ la_out_b,
            const float* __restrict__ ra_in_w,  const float* __restrict__ ra_in_b,
            const float* __restrict__ ra_out_w, const float* __restrict__ ra_out_b,
            float* __restrict__ cat) {
    __shared__ float bill[Dd];
    __shared__ int   idx_s[Ss];
    __shared__ float xn[Ss][Dd];
    __shared__ float q[Dd];
    __shared__ float tm[Hh][Dd];
    __shared__ float sb[Hh][Ss];
    __shared__ float u4[Hh][Dd];
    __shared__ float ov[Dd];

    const int b = blockIdx.x;
    const int d = threadIdx.x;
    bill[d] = x2[(long)query_idx[b] * Dd + d];
    cat[(long)b * E3 + Dd + d] = bill[d];
    __syncthreads();

    for (int side = 0; side < 2; ++side) {
        const int* idx            = (side ? subj_idx  : spon_idx)  + (long)b * Ss;
        const unsigned char* msk  = (side ? subj_mask : spon_mask) + (long)b * Ss;
        const float* Wi = side ? ra_in_w  : la_in_w;
        const float* bi = side ? ra_in_b  : la_in_b;
        const float* Wo = side ? ra_out_w : la_out_w;
        const float* bo = side ? ra_out_b : la_out_b;

        if (d < Ss) idx_s[d] = idx[d];
        __syncthreads();
        for (int j = 0; j < Ss; ++j) xn[j][d] = x2[(long)idx_s[j] * Dd + d];
        {   // q = Wq bill + bq   (rows 0..127 of in_proj)
            float acc = bi[d];
            const float* wr = Wi + (long)d * Dd;
            for (int k = 0; k < Dd; ++k) acc += wr[k] * bill[k];
            q[d] = acc;
        }
        __syncthreads();
        for (int h = 0; h < Hh; ++h) {      // tm[h] = Wk_h^T q_h  (rows 128..255)
            float t = 0.0f;
            for (int dd = 0; dd < 32; ++dd)
                t += q[h * 32 + dd] * Wi[(long)(Dd + h * 32 + dd) * Dd + d];
            tm[h][d] = t;
        }
        __syncthreads();
        if (d < Ss) {                       // scores + mask
            const float sc = 0.17677669529663687f;   // 1/sqrt(32)
            for (int h = 0; h < Hh; ++h) {
                float s = 0.0f;
                for (int e = 0; e < Dd; ++e) s += xn[d][e] * tm[h][e];
                s *= sc;
                if (msk[d]) s = -3.0e38f;
                sb[h][d] = s;
            }
        }
        __syncthreads();
        if (d < Hh) {                       // softmax over 50 keys
            float m = -3.0e38f;
            for (int j = 0; j < Ss; ++j) m = fmaxf(m, sb[d][j]);
            float den = 0.0f;
            for (int j = 0; j < Ss; ++j) den += expf(sb[d][j] - m);
            float inv = 1.0f / den;
            for (int j = 0; j < Ss; ++j) sb[d][j] = expf(sb[d][j] - m) * inv;
        }
        __syncthreads();
        for (int h = 0; h < Hh; ++h) {      // u_h = sum_j a[h,j] x_j
            float s = 0.0f;
            for (int j = 0; j < Ss; ++j) s += sb[h][j] * xn[j][d];
            u4[h][d] = s;
        }
        __syncthreads();
        {   // o[d] = Wv[d,:] . u_{head(d)} + bv[d]  (rows 256..383)
            int h = d >> 5;
            float s = bi[2 * Dd + d];
            const float* wv = Wi + (long)(2 * Dd + d) * Dd;
            for (int e = 0; e < Dd; ++e) s += wv[e] * u4[h][e];
            ov[d] = s;
        }
        __syncthreads();
        {   // out projection
            float s = bo[d];
            const float* wo = Wo + (long)d * Dd;
            for (int e = 0; e < Dd; ++e) s += wo[e] * ov[e];
            cat[(long)b * E3 + side * (2 * Dd) + d] = s;   // side0 -> left, side1 -> right
        }
        __syncthreads();
    }
}

// ---------------------------------------------------------------------------
// Kernel 4: generic WMMA GEMM  C[M,Nc] = A[M,K] @ W[Nc,K]^T + bias
// (fusion MLP; seq-len-1 self-attn softmax == 1, so only v/out/fc GEMMs live)
// Block tile 64x64, 8 waves, 2 tiles per wave.  K, Nc multiples of 32/64.
// ---------------------------------------------------------------------------
__global__ void __launch_bounds__(256)
gemm_wmma_kernel(const float* __restrict__ A, const float* __restrict__ W,
                 const float* __restrict__ bias, float* __restrict__ C,
                 int K, int Nc) {
    __shared__ __align__(32) _Float16 Af[4][32][16];
    __shared__ __align__(32) _Float16 Bf[4][32][16];
    const int tid  = threadIdx.x;
    const int lane = tid & 31;
    const int w    = tid >> 5;
    const int mb   = blockIdx.x * 64;
    const int nb   = blockIdx.y * 64;
    const int ms   = w & 3;
    const int ng   = w >> 2;
    v8f acc0 = {}, acc1 = {};

    const int p      = tid >> 1;
    const int hb     = (tid & 1) * 8;
    const int s_ms   = p >> 5;                 // A mstrip / B ntile
    const int s_lane = p & 31;
    const int s_row  = s_ms * 16 + (s_lane & 15);
    const int s_kb   = (s_lane < 16) ? 0 : 8;
    const int s_klo  = s_kb + (hb ? 16 : 0);

    for (int ks = 0; ks < K; ks += 32) {
        const float* ap = A + (long)(mb + s_row) * K + ks + s_klo;
        const float* wp = W + (long)(nb + s_row) * K + ks + s_klo;  // W row-major over k: contiguous
        #pragma unroll
        for (int j = 0; j < 8; ++j) {
            Af[s_ms][s_lane][hb + j] = (_Float16)ap[j];
            Bf[s_ms][s_lane][hb + j] = (_Float16)wp[j];
        }
        __syncthreads();
        v16h a  = *(const v16h*)&Af[ms][lane][0];
        v16h b0 = *(const v16h*)&Bf[ng * 2 + 0][lane][0];
        v16h b1 = *(const v16h*)&Bf[ng * 2 + 1][lane][0];
        acc0 = __builtin_amdgcn_wmma_f32_16x16x32_f16(false, a, false, b0, (short)0, acc0, false, false);
        acc1 = __builtin_amdgcn_wmma_f32_16x16x32_f16(false, a, false, b1, (short)0, acc1, false, false);
        __syncthreads();
    }
    const int col0 = nb + ng * 32 + (lane & 15);
    const int mrow = mb + ms * 16 + ((lane < 16) ? 0 : 8);
    #pragma unroll
    for (int i = 0; i < 8; ++i) {
        C[(long)(mrow + i) * Nc + col0]      = acc0[i] + bias[col0];
        C[(long)(mrow + i) * Nc + col0 + 16] = acc1[i] + bias[col0 + 16];
    }
}

// ---------------------------------------------------------------------------
// Kernel 5: scores[b] = fused128[b] . x2[leg[b]]  (one wave per bill)
// ---------------------------------------------------------------------------
__global__ void __launch_bounds__(256)
score_kernel(const float* __restrict__ f, const float* __restrict__ x2,
             const int* __restrict__ leg, float* __restrict__ out) {
    int wv   = (blockIdx.x * 256 + threadIdx.x) >> 5;
    int lane = threadIdx.x & 31;
    if (wv >= Bb) return;
    const float4 a = *(const float4*)(f  + (long)wv * Dd + lane * 4);
    const float4 b = *(const float4*)(x2 + (long)leg[wv] * Dd + lane * 4);
    float s = a.x * b.x + a.y * b.y + a.z * b.z + a.w * b.w;
    for (int off = 16; off > 0; off >>= 1) s += __shfl_down(s, off, 32);
    if (lane == 0) out[wv] = s;
}

// ---------------------------------------------------------------------------
// Launch
// ---------------------------------------------------------------------------
extern "C" void kernel_launch(void* const* d_in, const int* in_sizes, int n_in,
                              void* d_out, int out_size, void* d_ws, size_t ws_size,
                              hipStream_t stream) {
    const float* node_emb = (const float*)d_in[0];
    const float* W_rel    = (const float*)d_in[1];   // [2,8,128,128]
    const float* W_root   = (const float*)d_in[2];   // [2,128,128]
    const float* b_rgcn   = (const float*)d_in[3];   // [2,128]
    const float* la_in_w  = (const float*)d_in[4];
    const float* la_in_b  = (const float*)d_in[5];
    const float* la_out_w = (const float*)d_in[6];
    const float* la_out_b = (const float*)d_in[7];
    const float* ra_in_w  = (const float*)d_in[8];
    const float* ra_in_b  = (const float*)d_in[9];
    const float* ra_out_w = (const float*)d_in[10];
    const float* ra_out_b = (const float*)d_in[11];
    const float* fu_in_w  = (const float*)d_in[12];  // [1152,384]
    const float* fu_in_b  = (const float*)d_in[13];
    const float* fu_out_w = (const float*)d_in[14];  // [384,384]
    const float* fu_out_b = (const float*)d_in[15];
    const float* fc_w     = (const float*)d_in[16];  // [128,384]
    const float* fc_b     = (const float*)d_in[17];
    const int* edge_src   = (const int*)d_in[18];
    const int* edge_dst   = (const int*)d_in[19];
    const int* edge_type  = (const int*)d_in[20];
    const int* query_idx  = (const int*)d_in[21];
    const int* spon_idx   = (const int*)d_in[22];
    const int* subj_idx   = (const int*)d_in[23];
    const unsigned char* spon_mask = (const unsigned char*)d_in[24];
    const unsigned char* subj_mask = (const unsigned char*)d_in[25];
    const int* leg_idx    = (const int*)d_in[26];

    // workspace layout (~258 MB): x1 | x2 | cnt | agg (agg reused for fusion bufs)
    float* x1  = (float*)d_ws;
    float* x2  = x1  + (long)Nn * Dd;
    float* cnt = x2  + (long)Nn * Dd;
    float* agg = cnt + (long)Nn * Rr;
    float* cat   = agg;                                   // [B,384]   6.3 MB
    float* vbuf  = (float*)((char*)agg + (8u  << 20));    // [B,384]
    float* fused = (float*)((char*)agg + (16u << 20));    // [B,384]
    float* o128  = (float*)((char*)agg + (24u << 20));    // [B,128]

    const size_t aggBytes = (size_t)Nn * Rr * Dd * sizeof(float);
    const size_t cntBytes = (size_t)Nn * Rr * sizeof(float);
    const int scatterBlocks = (NEe * 32 + 255) / 256;
    const int gemmBlocks    = (Nn + 63) / 64;

    // ---- RGCN layer 0 ----
    hipMemsetAsync(agg, 0, aggBytes, stream);
    hipMemsetAsync(cnt, 0, cntBytes, stream);
    edge_scatter_kernel<<<scatterBlocks, 256, 0, stream>>>(node_emb, edge_src, edge_dst, edge_type, agg, cnt);
    rgcn_gemm_kernel<<<gemmBlocks, 256, 0, stream>>>(node_emb, agg, cnt,
        W_rel, W_root, b_rgcn, x1, /*relu*/1);

    // ---- RGCN layer 1 ----
    hipMemsetAsync(agg, 0, aggBytes, stream);
    hipMemsetAsync(cnt, 0, cntBytes, stream);
    edge_scatter_kernel<<<scatterBlocks, 256, 0, stream>>>(x1, edge_src, edge_dst, edge_type, agg, cnt);
    rgcn_gemm_kernel<<<gemmBlocks, 256, 0, stream>>>(x1, agg, cnt,
        W_rel + (long)Rr * Dd * Dd, W_root + (long)Dd * Dd, b_rgcn + Dd, x2, /*relu*/0);

    // ---- dual attention -> cat = [left | bill | right] ----
    attn_kernel<<<Bb, 128, 0, stream>>>(x2, query_idx, spon_idx, subj_idx,
        spon_mask, subj_mask,
        la_in_w, la_in_b, la_out_w, la_out_b,
        ra_in_w, ra_in_b, ra_out_w, ra_out_b, cat);

    // ---- fusion: v = cat @ Wv_fu^T + bv ; fused = v @ Wo_fu^T + bo ; o128 = fused @ fc^T + b ----
    gemm_wmma_kernel<<<dim3(Bb / 64, E3 / 64), 256, 0, stream>>>(
        cat, fu_in_w + (long)2 * E3 * E3, fu_in_b + 2 * E3, vbuf, E3, E3);
    gemm_wmma_kernel<<<dim3(Bb / 64, E3 / 64), 256, 0, stream>>>(
        vbuf, fu_out_w, fu_out_b, fused, E3, E3);
    gemm_wmma_kernel<<<dim3(Bb / 64, Dd / 64), 256, 0, stream>>>(
        fused, fc_w, fc_b, o128, E3, Dd);

    // ---- scores ----
    score_kernel<<<(Bb * 32 + 255) / 256, 256, 0, stream>>>(o128, x2, leg_idx, (float*)d_out);
}
